// GATv2_70875550319064
// MI455X (gfx1250) — matrix operand order repaired and checked by reading, not compiled, not run on hardware
//
#include <hip/hip_runtime.h>
#include <hip/hip_bf16.h>
#include <math.h>

// ---------------------------------------------------------------------------
// GATv2 two-layer GNN for MI455X (gfx1250, wave32, WMMA).
// N=50000, E=800000, Q=100000, IN=128, H*HID=128, H*OUT=128.
// ---------------------------------------------------------------------------

#define NNODES 50000
#define NEDGES 800000
#define NQUERY 100000
#define DIM    128      // all GEMM dims are 128 here
#define HEADS  4
#define HDIM   32       // per-head dim (both layers)
#define WPAD   136      // padded K-stride (bf16 elems) for LDS weight tile

typedef __attribute__((ext_vector_type(16))) __bf16        v16bf;
typedef __attribute__((ext_vector_type(8)))  float         v8f;
typedef __attribute__((ext_vector_type(4)))  float         v4f;
typedef __attribute__((ext_vector_type(4)))  unsigned int  v4u;

union BFrag { v16bf bf; v4u q[2]; unsigned short us[16]; };

__device__ __forceinline__ unsigned short f2bf(float f) {
  unsigned int u = __float_as_uint(f);
  unsigned int r = (u + 0x7FFFu + ((u >> 16) & 1u)) >> 16;   // RNE bf16
  return (unsigned short)r;
}

// float atomic-max via sign-aware integer atomics
// (global_atomic_max_i32 / global_atomic_min_u32; works with -inf init)
__device__ __forceinline__ void atomicMaxFloat(float* addr, float val) {
  if (val >= 0.0f) atomicMax((int*)addr, __float_as_int(val));
  else             atomicMin((unsigned int*)addr, (unsigned int)__float_as_int(val));
}

// ---------------------------------------------------------------------------
// Fill kernel (need -inf fills, so hipMemsetAsync is not enough)
// ---------------------------------------------------------------------------
__global__ void gat_fill_kernel(float* __restrict__ p, float v, size_t n) {
  size_t i = (size_t)blockIdx.x * blockDim.x + threadIdx.x;
  if (i < n) p[i] = v;
}

// ---------------------------------------------------------------------------
// Projection GEMM: out[N x 128] = act(in[N x 128]) @ W[128 x 128] + bias
// 256 threads = 8 waves; wave w owns rows [blk*128+16w, +16).
// Weights staged TRANSPOSED in LDS (bf16, padded stride) so every B fragment
// is two ds_load_b128 per lane.  A fragments are 4x global_load_b128 of the
// owning row.  32 v_wmma_f32_16x16x32_bf16 per wave, f32 accumulate.
// relu_in folds the inter-layer ReLU into the A load.
// ---------------------------------------------------------------------------
__global__ __launch_bounds__(256)
void gat_proj_wmma_kernel(const float* __restrict__ in,
                          const float* __restrict__ W,
                          const float* __restrict__ bias,
                          float* __restrict__ out,
                          int nrows, int relu_in) {
  // sWt[col][k] : transposed bf16 weights, K-stride padded to 136 (bank spread)
  __shared__ unsigned short sWt[DIM * WPAD];        // ~34 KB
  const int tid = threadIdx.x;
  for (int i = tid; i < DIM * DIM; i += 256) {
    const int k = i >> 7, n = i & 127;              // W row-major [k][n]
    sWt[n * WPAD + k] = f2bf(W[i]);
  }
  __syncthreads();

  const int waveid = tid >> 5;
  const int lane   = tid & 31;
  const int rowbase = blockIdx.x * 128 + waveid * 16;

  // ---- A-matrix mapping (16-bit A 16x32, ISA 7.12.2): both lane halves hold
  // rows M=0..15; K ranges differ by goff. Lane needs two contiguous 8-runs.
  const int arow  = rowbase + (lane & 15);
  const bool rowok = arow < nrows;
  const float* rowp = in + (size_t)arow * DIM;
  const int goff = (lane >> 4) * 8;

  // ---- B-matrix mapping (16-bit B 32x16): lane = N column; lane half picks
  // K sub-range.  With sWt transposed, a fragment = two 8-runs of K (+0, +16).
  const int bcol16 = lane & 15;
  const int bkoff  = (lane >> 4) * 8;

  v8f acc[8];
#pragma unroll
  for (int j = 0; j < 8; ++j) acc[j] = (v8f){0,0,0,0,0,0,0,0};

#pragma unroll
  for (int ks = 0; ks < 4; ++ks) {                  // K = 128 in 4 steps of 32
    BFrag a;
    if (rowok) {
      const int kb0 = ks * 32 + goff;               // run0: K = kb0..kb0+7
      const int kb1 = ks * 32 + 16 + goff;          // run1: K = kb1..kb1+7
      v4f x0 = *(const v4f*)(rowp + kb0);
      v4f x1 = *(const v4f*)(rowp + kb0 + 4);
      v4f x2 = *(const v4f*)(rowp + kb1);
      v4f x3 = *(const v4f*)(rowp + kb1 + 4);
      if (relu_in) {
#pragma unroll
        for (int c = 0; c < 4; ++c) {
          x0[c] = fmaxf(x0[c], 0.0f); x1[c] = fmaxf(x1[c], 0.0f);
          x2[c] = fmaxf(x2[c], 0.0f); x3[c] = fmaxf(x3[c], 0.0f);
        }
      }
#pragma unroll
      for (int c = 0; c < 4; ++c) {
        a.us[c]      = f2bf(x0[c]);
        a.us[4 + c]  = f2bf(x1[c]);
        a.us[8 + c]  = f2bf(x2[c]);
        a.us[12 + c] = f2bf(x3[c]);
      }
    } else {
#pragma unroll
      for (int c = 0; c < 16; ++c) a.us[c] = 0;
    }

#pragma unroll
    for (int j = 0; j < 8; ++j) {                   // 8 N-tiles -> 128 cols
      BFrag b;
      const unsigned short* bp =
          &sWt[(j * 16 + bcol16) * WPAD + ks * 32 + bkoff];
      b.q[0] = *(const v4u*)(bp);                   // K = +0..7   (ds_load_b128)
      b.q[1] = *(const v4u*)(bp + 16);              // K = +16..23 (ds_load_b128)
      acc[j] = __builtin_amdgcn_wmma_f32_16x16x32_bf16(
                   false, a.bf, false, b.bf, (short)0, acc[j], false, false);
    }
  }

  // ---- C/D layout: VGPR r -> M=r (lanes 0-15) / M=r+8 (lanes 16-31) ----
  const int col16 = lane & 15;
  const int rsel  = (lane >> 4) * 8;
#pragma unroll
  for (int j = 0; j < 8; ++j) {
    const int col = j * 16 + col16;
    const float bv = bias[col];
#pragma unroll
    for (int r = 0; r < 8; ++r) {
      const int row = rowbase + rsel + r;
      if (row < nrows) out[(size_t)row * DIM + col] = acc[j][r] + bv;
    }
  }
}

// ---------------------------------------------------------------------------
// Edge logits: one wave per edge.  Lane owns 4 contiguous dims (float4), so
// each row gather is one global_load_b128; head h = lane>>3 (8 lanes/head).
// logit[e][h] = sum_d attn[h][d] * leaky_relu(fs[src][h,d] + fd[dst][h,d])
// Folds segment-max via float atomic max.
// ---------------------------------------------------------------------------
__global__ __launch_bounds__(256)
void gat_edge_logits_kernel(const int* __restrict__ esrc,
                            const int* __restrict__ edst,
                            const float* __restrict__ fs,
                            const float* __restrict__ fd,
                            const float* __restrict__ attn,
                            float* __restrict__ logits,
                            float* __restrict__ maxbuf, int E) {
  const int e = blockIdx.x * (blockDim.x >> 5) + (threadIdx.x >> 5);
  const int lane = threadIdx.x & 31;
  if (e >= E) return;
  const int s = esrc[e], d = edst[e];
  const v4f va = *(const v4f*)(fs + (size_t)s * DIM + lane * 4);
  const v4f vb = *(const v4f*)(fd + (size_t)d * DIM + lane * 4);
  const v4f vw = *(const v4f*)(attn + lane * 4);

  float p = 0.0f;
#pragma unroll
  for (int c = 0; c < 4; ++c) {
    float v = va[c] + vb[c];
    v = v > 0.0f ? v : 0.2f * v;                    // leaky_relu(0.2)
    p += v * vw[c];
  }
  // reduce within each 8-lane head group (bits 0..2 only)
  p += __shfl_xor(p, 1, 32);
  p += __shfl_xor(p, 2, 32);
  p += __shfl_xor(p, 4, 32);

  if ((lane & 7) == 0) {
    const int h = lane >> 3;
    logits[(size_t)e * HEADS + h] = p;
    atomicMaxFloat(&maxbuf[(size_t)d * HEADS + h], p);
  }
}

// ---------------------------------------------------------------------------
// exp(logit - segmax) in-place + segment-sum denominators.
// One thread per (edge, head).
// ---------------------------------------------------------------------------
__global__ __launch_bounds__(256)
void gat_edge_exp_kernel(const int* __restrict__ edst,
                         float* __restrict__ logits,      // in: logit, out: exp
                         const float* __restrict__ maxbuf,
                         float* __restrict__ denom, int E) {
  const int t = blockIdx.x * blockDim.x + threadIdx.x;
  if (t >= E * HEADS) return;
  const int e = t >> 2, h = t & (HEADS - 1);
  const int d = edst[e];
  const float ex = __expf(logits[t] - maxbuf[(size_t)d * HEADS + h]);
  logits[t] = ex;
  atomicAdd(&denom[(size_t)d * HEADS + h], ex);
}

// ---------------------------------------------------------------------------
// Aggregation: one wave per edge, lane owns 4 contiguous dims (b128 gather),
// alpha = ex/denom[dst]; scatter alpha * fs[src] with 4 contiguous f32 atomics.
// ---------------------------------------------------------------------------
__global__ __launch_bounds__(256)
void gat_edge_agg_kernel(const int* __restrict__ esrc,
                         const int* __restrict__ edst,
                         const float* __restrict__ fs,
                         const float* __restrict__ exbuf,
                         const float* __restrict__ denom,
                         float* __restrict__ agg, int E) {
  const int e = blockIdx.x * (blockDim.x >> 5) + (threadIdx.x >> 5);
  const int lane = threadIdx.x & 31;
  if (e >= E) return;
  const int s = esrc[e], d = edst[e];
  const int h = lane >> 3;                          // head owned by this lane
  const v4f va = *(const v4f*)(fs + (size_t)s * DIM + lane * 4);
  const float alpha = exbuf[(size_t)e * HEADS + h] /
                      denom[(size_t)d * HEADS + h];
  float* pa = agg + (size_t)d * DIM + lane * 4;
#pragma unroll
  for (int c = 0; c < 4; ++c) atomicAdd(pa + c, alpha * va[c]);
}

// ---------------------------------------------------------------------------
// Link-prediction scores: one wave per query, 128-dim dot (b128 gathers),
// full wave reduction, sigmoid.
// ---------------------------------------------------------------------------
__global__ __launch_bounds__(256)
void gat_score_kernel(const int* __restrict__ qsrc,
                      const int* __restrict__ qdst,
                      const float* __restrict__ h2,
                      float* __restrict__ out, int Q) {
  const int q = blockIdx.x * (blockDim.x >> 5) + (threadIdx.x >> 5);
  const int lane = threadIdx.x & 31;
  if (q >= Q) return;
  const v4f va = *(const v4f*)(h2 + (size_t)qsrc[q] * DIM + lane * 4);
  const v4f vb = *(const v4f*)(h2 + (size_t)qdst[q] * DIM + lane * 4);
  float p = va[0] * vb[0] + va[1] * vb[1] + va[2] * vb[2] + va[3] * vb[3];
#pragma unroll
  for (int m = 16; m >= 1; m >>= 1) p += __shfl_xor(p, m, 32);
  if (lane == 0) out[q] = 1.0f / (1.0f + __expf(-p));
}

// ---------------------------------------------------------------------------
// Launcher
// ---------------------------------------------------------------------------
extern "C" void kernel_launch(void* const* d_in, const int* in_sizes, int n_in,
                              void* d_out, int out_size, void* d_ws, size_t ws_size,
                              hipStream_t stream) {
  (void)in_sizes; (void)n_in; (void)out_size; (void)ws_size;

  const float* feat  = (const float*)d_in[0];
  const int*   esrc  = (const int*)  d_in[1];
  const int*   edst  = (const int*)  d_in[2];
  const int*   qsrc  = (const int*)  d_in[3];
  const int*   qdst  = (const int*)  d_in[4];
  const float* W1s   = (const float*)d_in[5];
  const float* b1s   = (const float*)d_in[6];
  const float* W1d   = (const float*)d_in[7];
  const float* b1d   = (const float*)d_in[8];
  const float* attn1 = (const float*)d_in[9];
  const float* W2s   = (const float*)d_in[10];
  const float* b2s   = (const float*)d_in[11];
  const float* W2d   = (const float*)d_in[12];
  const float* b2d   = (const float*)d_in[13];
  const float* attn2 = (const float*)d_in[14];

  const int N = NNODES, E = NEDGES, Q = NQUERY;

  // Workspace layout (floats). fs/fd reused across layers; h buffer holds h1,
  // then is recycled for h2 after the layer-2 projections have consumed h1.
  float* ws = (float*)d_ws;
  float* fs = ws;                         // N*128
  float* fd = fs + (size_t)N * DIM;       // N*128
  float* h  = fd + (size_t)N * DIM;       // N*128
  float* ex = h  + (size_t)N * DIM;       // E*4  (logits, then exp, in place)
  float* mx = ex + (size_t)E * HEADS;     // N*4
  float* dn = mx + (size_t)N * HEADS;     // N*4

  const dim3 blk(256);
  const int projBlocks = (N + 127) / 128;
  const int edgeBlocks = (E + 7) / 8;            // 8 waves (edges) per block
  const int expBlocks  = (E * HEADS + 255) / 256;
  const int qBlocks    = (Q + 7) / 8;
  const float ninf = -__builtin_inff();

  const float* Wsrc[2]  = {W1s, W2s};
  const float* bsrc[2]  = {b1s, b2s};
  const float* Wdst[2]  = {W1d, W2d};
  const float* bdstp[2] = {b1d, b2d};
  const float* attn[2]  = {attn1, attn2};

  for (int layer = 0; layer < 2; ++layer) {
    const float* input  = (layer == 0) ? feat : h;
    const int    reluIn = (layer == 0) ? 0 : 1;    // fold inter-layer ReLU

    gat_proj_wmma_kernel<<<projBlocks, blk, 0, stream>>>(
        input, Wsrc[layer], bsrc[layer], fs, N, reluIn);
    gat_proj_wmma_kernel<<<projBlocks, blk, 0, stream>>>(
        input, Wdst[layer], bdstp[layer], fd, N, reluIn);

    gat_fill_kernel<<<(N * HEADS + 255) / 256, blk, 0, stream>>>(mx, ninf, (size_t)N * HEADS);
    gat_fill_kernel<<<(N * HEADS + 255) / 256, blk, 0, stream>>>(dn, 0.0f, (size_t)N * HEADS);

    gat_edge_logits_kernel<<<edgeBlocks, blk, 0, stream>>>(
        esrc, edst, fs, fd, attn[layer], ex, mx, E);
    gat_edge_exp_kernel<<<expBlocks, blk, 0, stream>>>(edst, ex, mx, dn, E);

    // h1 already consumed by the layer-2 projections above; safe to recycle.
    gat_fill_kernel<<<((size_t)N * DIM + 255) / 256, blk, 0, stream>>>(h, 0.0f, (size_t)N * DIM);
    gat_edge_agg_kernel<<<edgeBlocks, blk, 0, stream>>>(esrc, edst, fs, ex, dn, h, E);
  }

  gat_score_kernel<<<qBlocks, blk, 0, stream>>>(qsrc, qdst, h, (float*)d_out, Q);
}
